// LossFunction_86870008529065
// MI455X (gfx1250) — matrix operand order repaired and checked by reading, not compiled, not run on hardware
//
#include <hip/hip_runtime.h>
#include <math.h>

#define BATCH   512
#define NUMC    100000
#define KDIM    512
#define S_SCALE 30.0f
#define NTILE_N 6250     // 100000 / 16
#define NBX     196      // ceil(6250 / 32) class-chunks of 512

typedef _Float16 v16h __attribute__((ext_vector_type(16)));
typedef _Float16 v8h  __attribute__((ext_vector_type(8)));
typedef float    v8f  __attribute__((ext_vector_type(8)));
typedef float    v4f  __attribute__((ext_vector_type(4)));

// ---------------------------------------------------------------------------
// Kernel 1: per-sample norm, unit-x in f16, margin params, MagFace g(a)
// ---------------------------------------------------------------------------
__global__ __launch_bounds__(256)
void prep_x_kernel(const float* __restrict__ x, _Float16* __restrict__ xf16,
                   float* __restrict__ cosm, float* __restrict__ sinm,
                   float* __restrict__ mmv, float* __restrict__ grow) {
    const int row = blockIdx.x;
    const int tid = threadIdx.x;
    const float* xr = x + (size_t)row * KDIM;
    float v0 = xr[tid];
    float v1 = xr[tid + 256];
    float ss = v0 * v0 + v1 * v1;
    #pragma unroll
    for (int m = 16; m >= 1; m >>= 1) ss += __shfl_xor(ss, m, 32);
    __shared__ float wsum[8];
    if ((tid & 31) == 0) wsum[tid >> 5] = ss;
    __syncthreads();
    float tot = 0.f;
    #pragma unroll
    for (int i = 0; i < 8; ++i) tot += wsum[i];
    float nrm = sqrtf(tot);
    float inv = 1.0f / nrm;
    xf16[(size_t)row * KDIM + tid]       = (_Float16)(v0 * inv);
    xf16[(size_t)row * KDIM + tid + 256] = (_Float16)(v1 * inv);
    if (tid == 0) {
        float nc  = fminf(110.0f, fmaxf(10.0f, nrm));
        float ada = (1.0f - 0.1f) / (110.0f - 10.0f) * (nc - 10.0f) + 0.1f;
        cosm[row] = cosf(ada);
        sinm[row] = sinf(ada);
        mmv[row]  = sinf(ada) * ada;               // sin(pi-m)*m == sin(m)*m
        grow[row] = nc / (110.0f * 110.0f) + 1.0f / nc;
    }
}

// ---------------------------------------------------------------------------
// Kernel 2: inverse L2 norm per class row; if PRECONV also emit unit-norm f16
// weight rows (f32 source read non-temporally: it is dead after this pass,
// and we want the f16 copy, which fits the 192MB L2, to stay resident).
// ---------------------------------------------------------------------------
template<bool PRECONV>
__global__ __launch_bounds__(256)
void wnorm_kernel(const float* __restrict__ wgt, float* __restrict__ inv_wn,
                  _Float16* __restrict__ wf16) {
    const int tid  = threadIdx.x;
    const int lane = tid & 31;
    const int row  = blockIdx.x * 8 + (tid >> 5);
    const float* wr = wgt + (size_t)row * KDIM + lane * 16;
    v4f v[4];
    float ss = 0.f;
    #pragma unroll
    for (int j = 0; j < 4; ++j) {
        v[j] = PRECONV ? __builtin_nontemporal_load((const v4f*)(wr) + j)
                       : *((const v4f*)(wr) + j);
        ss += v[j][0]*v[j][0] + v[j][1]*v[j][1] + v[j][2]*v[j][2] + v[j][3]*v[j][3];
    }
    #pragma unroll
    for (int m = 16; m >= 1; m >>= 1) ss += __shfl_xor(ss, m, 32);
    float inv = 1.0f / sqrtf(ss);
    if (lane == 0) inv_wn[row] = inv;
    if (PRECONV) {
        v16h h;
        #pragma unroll
        for (int j = 0; j < 4; ++j) {
            #pragma unroll
            for (int e = 0; e < 4; ++e) h[j * 4 + e] = (_Float16)(v[j][e] * inv);
        }
        *(v16h*)(wf16 + (size_t)row * KDIM + lane * 16) = h;
    }
}

// ---------------------------------------------------------------------------
// Kernel 3: WMMA GEMM (f16 in, f32 acc) + fused margin/softmax-partial epilogue
//   grid = (NBX, 16): blockIdx.x -> 512-class chunk, blockIdx.y -> 32-row strip
//   4 waves/block; each wave: 8 n-tiles x 2 m-tiles, full K=512.
//   B fragment loaded once per k-step, reused by two WMMAs.
// ---------------------------------------------------------------------------
template<bool PRECONV>
__global__ __launch_bounds__(128)
void magface_gemm_kernel(const float* __restrict__ wgt, const _Float16* __restrict__ wf16,
                         const int* __restrict__ label,
                         const _Float16* __restrict__ xf16, const float* __restrict__ inv_wn,
                         const float* __restrict__ cosm, const float* __restrict__ sinm,
                         const float* __restrict__ mmv,
                         float* __restrict__ tlogit,
                         float* __restrict__ psum, float* __restrict__ pmax,
                         int* __restrict__ pidx) {
    const int tid     = threadIdx.x;
    const int lane    = tid & 31;
    const int wv      = tid >> 5;      // wave 0..3
    const int col     = lane & 15;     // N within tile (also M for A-rows)
    const int lh      = lane >> 4;     // lane-half selects K slice
    const int bx      = blockIdx.x;
    const int m_base0 = blockIdx.y * 32;

    // A fragments: rows m_base0+col and m_base0+16+col; K pattern 0..15 + 16*lh
    const _Float16* xrow0 = xf16 + (size_t)(m_base0 + col) * KDIM;
    const _Float16* xrow1 = xrow0 + (size_t)16 * KDIM;
    const int ka = lh * 16;
    const int kb = lh * 16;            // B fragment: contiguous 16 K-values

    int labv[16];
    #pragma unroll
    for (int mt = 0; mt < 2; ++mt)
        #pragma unroll
        for (int i = 0; i < 8; ++i)
            labv[mt * 8 + i] = label[m_base0 + mt * 16 + i + lh * 8];

    float rsum[16], rmax[16]; int ridx[16];
    #pragma unroll
    for (int i = 0; i < 16; ++i) { rsum[i] = 0.f; rmax[i] = -3.0e38f; ridx[i] = 0x7fffffff; }

    for (int nt = 0; nt < 8; ++nt) {
        const int n_tile = bx * 32 + wv * 8 + nt;
        if (n_tile < NTILE_N) {
            const int c = n_tile * 16 + col;           // this lane's class column
            float invw = 1.0f;
            const float*    wrow32 = nullptr;
            const _Float16* wrow16 = nullptr;
            if (PRECONV) wrow16 = wf16 + (size_t)c * KDIM;
            else        { wrow32 = wgt  + (size_t)c * KDIM; invw = inv_wn[c]; }

            v8f acc[2];
            #pragma unroll
            for (int i = 0; i < 8; ++i) { acc[0][i] = 0.f; acc[1][i] = 0.f; }

            #pragma unroll 2
            for (int k0 = 0; k0 < KDIM; k0 += 32) {
                v16h bv;
                if (PRECONV) {
                    bv = *(const v16h*)(wrow16 + k0 + kb);      // 32B contiguous
                } else {
                    const float* bp = wrow32 + k0 + kb;
                    v4f b0 = *(const v4f*)(bp);
                    v4f b1 = *(const v4f*)(bp + 4);
                    v4f b2 = *(const v4f*)(bp + 8);
                    v4f b3 = *(const v4f*)(bp + 12);
                    #pragma unroll
                    for (int i = 0; i < 4; ++i) {
                        bv[i]      = (_Float16)b0[i];
                        bv[i + 4]  = (_Float16)b1[i];
                        bv[i + 8]  = (_Float16)b2[i];
                        bv[i + 12] = (_Float16)b3[i];
                    }
                }
                v16h av0, av1;
                {
                    const _Float16* ap = xrow0 + k0 + ka;
                    v8h alo = *(const v8h*)(ap);
                    v8h ahi = *(const v8h*)(ap + 8);
                    #pragma unroll
                    for (int i = 0; i < 8; ++i) { av0[i] = alo[i]; av0[i + 8] = ahi[i]; }
                }
                {
                    const _Float16* ap = xrow1 + k0 + ka;
                    v8h alo = *(const v8h*)(ap);
                    v8h ahi = *(const v8h*)(ap + 8);
                    #pragma unroll
                    for (int i = 0; i < 8; ++i) { av1[i] = alo[i]; av1[i + 8] = ahi[i]; }
                }
                acc[0] = __builtin_amdgcn_wmma_f32_16x16x32_f16(
                            false, av0, false, bv, (short)0, acc[0], false, false);
                acc[1] = __builtin_amdgcn_wmma_f32_16x16x32_f16(
                            false, av1, false, bv, (short)0, acc[1], false, false);
            }

            // Epilogue: C layout => VGPR i holds M=i (lanes 0-15) / M=i+8 (16-31)
            #pragma unroll
            for (int mt = 0; mt < 2; ++mt) {
                #pragma unroll
                for (int i = 0; i < 8; ++i) {
                    const int idx = mt * 8 + i;
                    float cosv  = PRECONV ? acc[mt][i] : acc[mt][i] * invw;
                    float logit = S_SCALE * cosv;
                    if (c == labv[idx]) {
                        int mg = m_base0 + mt * 16 + i + lh * 8;
                        float cm = cosm[mg], sm = sinm[mg], mv = mmv[mg];
                        float sine = sqrtf(fmaxf(0.f, 1.f - cosv * cosv));
                        float phi  = cosv * cm - sine * sm;
                        // cosine - th > 0  with th = -cos_m  =>  cosv + cm > 0
                        logit = S_SCALE * (((cosv + cm) > 0.f) ? phi : (cosv - mv));
                        tlogit[mg] = logit;             // unique writer per row
                    }
                    float e = __expf(logit - 30.0f);    // logits <= 30 always
                    rsum[idx] += e;
                    if (logit > rmax[idx] || (logit == rmax[idx] && c < ridx[idx])) {
                        rmax[idx] = logit; ridx[idx] = c;
                    }
                }
            }
        }
    }

    // reduce across the 16 lanes holding each row's columns (stays in half)
    #pragma unroll
    for (int i = 0; i < 16; ++i) {
        #pragma unroll
        for (int msk = 8; msk >= 1; msk >>= 1) {
            float os = __shfl_xor(rsum[i], msk, 32);
            float om = __shfl_xor(rmax[i], msk, 32);
            int   oi = __shfl_xor(ridx[i], msk, 32);
            rsum[i] += os;
            if (om > rmax[i] || (om == rmax[i] && oi < ridx[i])) { rmax[i] = om; ridx[i] = oi; }
        }
    }

    __shared__ float ls_sum[4][32];
    __shared__ float ls_max[4][32];
    __shared__ int   ls_idx[4][32];
    if (col == 0) {
        #pragma unroll
        for (int mt = 0; mt < 2; ++mt) {
            #pragma unroll
            for (int i = 0; i < 8; ++i) {
                int r = mt * 16 + i + lh * 8;
                ls_sum[wv][r] = rsum[mt * 8 + i];
                ls_max[wv][r] = rmax[mt * 8 + i];
                ls_idx[wv][r] = ridx[mt * 8 + i];
            }
        }
    }
    __syncthreads();
    if (tid < 32) {
        float s = 0.f, mx = -3.0e38f; int ix = 0x7fffffff;
        #pragma unroll
        for (int w2 = 0; w2 < 4; ++w2) {
            s += ls_sum[w2][tid];
            float om = ls_max[w2][tid]; int oi = ls_idx[w2][tid];
            if (om > mx || (om == mx && oi < ix)) { mx = om; ix = oi; }
        }
        int mg = m_base0 + tid;
        psum[(size_t)mg * NBX + bx] = s;
        pmax[(size_t)mg * NBX + bx] = mx;
        pidx[(size_t)mg * NBX + bx] = ix;
    }
}

// ---------------------------------------------------------------------------
// Kernel 4: per-row reduction over NBX chunks -> CE term + top-1 hit
// ---------------------------------------------------------------------------
__global__ __launch_bounds__(256)
void row_reduce_kernel(const float* __restrict__ psum, const float* __restrict__ pmax,
                       const int* __restrict__ pidx, const float* __restrict__ tlogit,
                       const int* __restrict__ label,
                       float* __restrict__ cerow, float* __restrict__ accrow) {
    const int m = blockIdx.x;
    const int tid = threadIdx.x;
    __shared__ float ss[256]; __shared__ float sm[256]; __shared__ int si[256];
    float s = 0.f, mx = -3.0e38f; int ix = 0x7fffffff;
    if (tid < NBX) {
        s  = psum[(size_t)m * NBX + tid];
        mx = pmax[(size_t)m * NBX + tid];
        ix = pidx[(size_t)m * NBX + tid];
    }
    ss[tid] = s; sm[tid] = mx; si[tid] = ix;
    __syncthreads();
    for (int st = 128; st > 0; st >>= 1) {
        if (tid < st) {
            ss[tid] += ss[tid + st];
            float om = sm[tid + st]; int oi = si[tid + st];
            if (om > sm[tid] || (om == sm[tid] && oi < si[tid])) { sm[tid] = om; si[tid] = oi; }
        }
        __syncthreads();
    }
    if (tid == 0) {
        // log_softmax target = tlogit - (30 + log(sum exp(l-30)))
        cerow[m]  = 30.0f + logf(ss[0]) - tlogit[m];
        accrow[m] = (si[0] == label[m]) ? 1.0f : 0.0f;
    }
}

// ---------------------------------------------------------------------------
// Kernel 5: final scalar reduction -> d_out[0]=loss, d_out[1]=prec1
// ---------------------------------------------------------------------------
__global__ __launch_bounds__(512)
void final_kernel(const float* __restrict__ cerow, const float* __restrict__ accrow,
                  const float* __restrict__ grow, float* __restrict__ out) {
    const int tid = threadIdx.x;
    __shared__ float a[512]; __shared__ float b[512]; __shared__ float g[512];
    a[tid] = cerow[tid]; b[tid] = accrow[tid]; g[tid] = grow[tid];
    __syncthreads();
    for (int st = 256; st > 0; st >>= 1) {
        if (tid < st) { a[tid] += a[tid + st]; b[tid] += b[tid + st]; g[tid] += g[tid + st]; }
        __syncthreads();
    }
    if (tid == 0) {
        out[0] = a[0] / 512.0f + 35.0f * (g[0] / 512.0f);
        out[1] = b[0] * (100.0f / 512.0f);
    }
}

// ---------------------------------------------------------------------------
extern "C" void kernel_launch(void* const* d_in, const int* in_sizes, int n_in,
                              void* d_out, int out_size, void* d_ws, size_t ws_size,
                              hipStream_t stream) {
    const float* x     = (const float*)d_in[0];
    const int*   label = (const int*)d_in[1];
    const float* wgt   = (const float*)d_in[2];
    float* out = (float*)d_out;

    char* p = (char*)d_ws;
    auto alloc = [&](size_t bytes) -> char* {
        char* r = p; p += (bytes + 511) & ~(size_t)511; return r;
    };
    _Float16* xf16   = (_Float16*)alloc((size_t)BATCH * KDIM * sizeof(_Float16));
    float*    inv_wn = (float*)alloc((size_t)NUMC * sizeof(float));
    float*    cosm   = (float*)alloc(BATCH * sizeof(float));
    float*    sinm   = (float*)alloc(BATCH * sizeof(float));
    float*    mmv    = (float*)alloc(BATCH * sizeof(float));
    float*    grow   = (float*)alloc(BATCH * sizeof(float));
    float*    tlog   = (float*)alloc(BATCH * sizeof(float));
    float*    cerow  = (float*)alloc(BATCH * sizeof(float));
    float*    accrow = (float*)alloc(BATCH * sizeof(float));
    float*    psum   = (float*)alloc((size_t)BATCH * NBX * sizeof(float));
    float*    pmax   = (float*)alloc((size_t)BATCH * NBX * sizeof(float));
    int*      pidx   = (int*)alloc((size_t)BATCH * NBX * sizeof(int));
    _Float16* wf16   = (_Float16*)alloc((size_t)NUMC * KDIM * sizeof(_Float16));

    // Use the pre-converted unit-norm f16 weight path only if workspace allows.
    const bool preconv = ((size_t)(p - (char*)d_ws) <= ws_size);

    prep_x_kernel<<<BATCH, 256, 0, stream>>>(x, xf16, cosm, sinm, mmv, grow);
    dim3 g3(NBX, BATCH / 32);
    if (preconv) {
        wnorm_kernel<true><<<NUMC / 8, 256, 0, stream>>>(wgt, inv_wn, wf16);
        magface_gemm_kernel<true><<<g3, 128, 0, stream>>>(wgt, wf16, label, xf16, inv_wn,
                                                          cosm, sinm, mmv, tlog,
                                                          psum, pmax, pidx);
    } else {
        wnorm_kernel<false><<<NUMC / 8, 256, 0, stream>>>(wgt, inv_wn, wf16);
        magface_gemm_kernel<false><<<g3, 128, 0, stream>>>(wgt, wf16, label, xf16, inv_wn,
                                                           cosm, sinm, mmv, tlog,
                                                           psum, pmax, pidx);
    }
    row_reduce_kernel<<<BATCH, 256, 0, stream>>>(psum, pmax, pidx, tlog, label,
                                                 cerow, accrow);
    final_kernel<<<1, 512, 0, stream>>>(cerow, accrow, grow, out);
}
